// CrossAttention_20504173871333
// MI455X (gfx1250) — compile-verified, matrix operand order checked by reading, not run on hardware
//
#include <hip/hip_runtime.h>

typedef __attribute__((ext_vector_type(16))) _Float16 v16h;
typedef __attribute__((ext_vector_type(8)))  float    v8f;

#define DH    128
#define LSEQ  8192
#define LOG2E 1.44269504088896340736f
#define SCALE 0.08838834764831845f   /* 1/sqrt(128) */

__global__ __launch_bounds__(256) void cross_attn_wmma_kernel(
    const float* __restrict__ x1, const float* __restrict__ x2,
    const float* __restrict__ Wq, const float* __restrict__ Wk,
    float* __restrict__ out)
{
    __shared__ float s_wk[DH];
    __shared__ float s_x2[DH];
    __shared__ float s_prod[DH];
    __shared__ float s_qs, s_maxWk, s_minWk;
    __shared__ float s_nd[8][32];   // per wave: [0..15]=num rows, [16..31]=den rows

    const int b    = blockIdx.x;
    const int tid  = threadIdx.x;
    const int wave = tid >> 5;
    const int lane = tid & 31;

    // ---- stage Wk, x2[b,:], Wq*x2 products in LDS ----
    if (tid < DH) {
        float wkv = Wk[tid];
        float x2v = x2[b * DH + tid];
        s_wk[tid]   = wkv;
        s_x2[tid]   = x2v;
        s_prod[tid] = Wq[tid] * x2v;
    }
    __syncthreads();
    if (tid == 0) {
        float qk = 0.f, mx = s_wk[0], mn = s_wk[0];
        for (int j = 0; j < DH; ++j) {
            qk += s_prod[j];
            mx  = fmaxf(mx, s_wk[j]);
            mn  = fminf(mn, s_wk[j]);
        }
        s_qs    = qk * SCALE * LOG2E;   // fold scale + log2(e) into the per-row scalar
        s_maxWk = mx;
        s_minWk = mn;
    }
    __syncthreads();

    const float qs = s_qs, mxw = s_maxWk, mnw = s_minWk;

    const int row    = lane & 15;     // M (for A) / N (for B)
    const int laneHi = lane >> 4;
    const int i0     = blockIdx.y * 128 + wave * 16;

    // per-row scalar in log2 domain; row-max is analytic (rank-1 scores)
    const float x1v = x1[(size_t)b * LSEQ + i0 + row];
    const float t   = x1v * qs;
    const float mt  = fmaxf(t * mxw, t * mnw);

    // ---- build B (32x16 f16) once: col0 = x2[b,j], col1 = 1, else 0 ----
    // lane = column N, K striped across packed VGPR halves (mirror of A layout)
    v16h Bm[4];
#pragma unroll
    for (int c = 0; c < 4; ++c) {
#pragma unroll
        for (int v = 0; v < 8; ++v) {
            int kb = ((v < 4) ? 0 : 16) + (laneHi ? 8 : 0) + 2 * (v & 3);
            int j  = 32 * c + kb;
            float b0 = (row == 0) ? s_x2[j]     : ((row == 1) ? 1.f : 0.f);
            float b1 = (row == 0) ? s_x2[j + 1] : ((row == 1) ? 1.f : 0.f);
            Bm[c][2 * v]     = (_Float16)b0;
            Bm[c][2 * v + 1] = (_Float16)b1;
        }
    }

    // ---- A = exp tile, generated directly in WMMA A-layout; 4 chained WMMAs ----
    v8f acc = {};
#pragma unroll
    for (int c = 0; c < 4; ++c) {
        v16h A;
#pragma unroll
        for (int v = 0; v < 8; ++v) {
            int kb = ((v < 4) ? 0 : 16) + (laneHi ? 8 : 0) + 2 * (v & 3);
            int j  = 32 * c + kb;
            float e0 = __builtin_amdgcn_exp2f(fmaf(t, s_wk[j],     -mt));
            float e1 = __builtin_amdgcn_exp2f(fmaf(t, s_wk[j + 1], -mt));
            A[2 * v]     = (_Float16)e0;
            A[2 * v + 1] = (_Float16)e1;
        }
        acc = __builtin_amdgcn_wmma_f32_16x16x32_f16(
            /*neg_a=*/false, A, /*neg_b=*/false, Bm[c],
            /*c_mod=*/(short)0, acc, /*reuse_a=*/false, /*reuse_b=*/false);
    }

    // ---- epilogue: C tile N=0 -> numerators, N=1 -> denominators ----
    float* nd = s_nd[wave];
    if (lane == 0) {
#pragma unroll
        for (int r = 0; r < 8; ++r) nd[r] = acc[r];          // rows 0..7 num
    } else if (lane == 16) {
#pragma unroll
        for (int r = 0; r < 8; ++r) nd[8 + r] = acc[r];      // rows 8..15 num
    } else if (lane == 1) {
#pragma unroll
        for (int r = 0; r < 8; ++r) nd[16 + r] = acc[r];     // rows 0..7 den
    } else if (lane == 17) {
#pragma unroll
        for (int r = 0; r < 8; ++r) nd[24 + r] = acc[r];     // rows 8..15 den
    }
    __syncthreads();
    if (lane < 16) {
        out[(size_t)b * LSEQ + i0 + lane] = nd[lane] / nd[16 + lane];
    }
}

extern "C" void kernel_launch(void* const* d_in, const int* in_sizes, int n_in,
                              void* d_out, int out_size, void* d_ws, size_t ws_size,
                              hipStream_t stream) {
    const float* x1 = (const float*)d_in[0];   // [128, 8192]
    const float* x2 = (const float*)d_in[1];   // [128, 128]
    const float* Wq = (const float*)d_in[2];   // [128]
    const float* Wk = (const float*)d_in[3];   // [128]
    float* out = (float*)d_out;                // [128, 8192]

    dim3 grid(128, LSEQ / 128);   // (b, 128-row slab) ; each of 8 waves owns one 16-row tile
    dim3 block(256);
    cross_attn_wmma_kernel<<<grid, block, 0, stream>>>(x1, x2, Wq, Wk, out);
}
